// GraphSAGE_25537875542171
// MI455X (gfx1250) — matrix operand order repaired and checked by reading, not compiled, not run on hardware
//
#include <hip/hip_runtime.h>
#include <hip/hip_bf16.h>
#include <stdint.h>

#define HID 64
#define NGRAPH 64

typedef __attribute__((ext_vector_type(16))) __bf16 v16bf;
typedef __attribute__((ext_vector_type(8)))  float  v8f;

struct Frag32B { uint4 lo; uint4 hi; };   // 32 bytes == v16bf

static __device__ __forceinline__ unsigned short f32_to_bf16_bits(float f) {
    union { float f; uint32_t u; } v; v.f = f;
    uint32_t u = v.u;
    uint32_t r = u + 0x7FFFu + ((u >> 16) & 1u);   // round-to-nearest-even
    return (unsigned short)(r >> 16);
}

static __device__ __forceinline__ unsigned int pack2(float a, float b) {
    return (unsigned int)f32_to_bf16_bits(a) | ((unsigned int)f32_to_bf16_bits(b) << 16);
}

// ---------------------------------------------------------------- utilities
__global__ void k_zero(float* __restrict__ p, int n) {
    int i = blockIdx.x * blockDim.x + threadIdx.x;
    if (i < n) p[i] = 0.0f;
}

// Convert 64x64 f32 weight [K][n] to bf16, swizzled into WMMA B-fragment order:
// dst[ (((K>>5)*64 + n)*2 + ((K>>4)&1))*16 + (K&15) ]
// so a lane's 16 B-fragment elements (fixed k0, half, n) are 32 contiguous bytes.
__global__ void k_w_to_bf16_swz(const float* __restrict__ src,
                                unsigned short* __restrict__ dst, int n) {
    int i = blockIdx.x * blockDim.x + threadIdx.x;
    if (i >= n) return;
    int K = i >> 6;
    int c = i & 63;
    int d = (((K >> 5) * 64 + c) * 2 + ((K >> 4) & 1)) * 16 + (K & 15);
    dst[d] = f32_to_bf16_bits(src[i]);
}

// ------------------------------------------------- layer 1 (input dim == 1)
__global__ void k_edge_deg_sx(const int* __restrict__ esrc, const int* __restrict__ edst,
                              const float* __restrict__ x,
                              float* __restrict__ sx, float* __restrict__ deg, int E) {
    int e = blockIdx.x * blockDim.x + threadIdx.x;
    if (e >= E) return;
    int s = esrc[e], d = edst[e];
    atomicAdd(&sx[d], x[s]);
    atomicAdd(&deg[d], 1.0f);
}

__global__ void k_layer1(const float* __restrict__ x, const float* __restrict__ sx,
                         const float* __restrict__ deg,
                         const float* __restrict__ W1l, const float* __restrict__ b1,
                         const float* __restrict__ W1r, float* __restrict__ h, int N) {
    int n = blockIdx.x;
    int c = threadIdx.x;
    if (n >= N) return;
    float mean = sx[n] / fmaxf(deg[n], 1.0f);
    float xv = x[n];
    float v = mean * W1l[c] + b1[c] + xv * W1r[c];
    h[n * HID + c] = fmaxf(v, 0.0f) + xv;
}

// ---------------------------------------------------- edge scatter (64-dim)
__global__ void k_aggregate(const int* __restrict__ esrc, const int* __restrict__ edst,
                            const float* __restrict__ h, float* __restrict__ agg, int E) {
    int gid = blockIdx.x * blockDim.x + threadIdx.x;
    int total = E * 16;
    if (gid >= total) return;
    int e = gid >> 4;
    int q = gid & 15;
    int s = esrc[e], d = edst[e];
    float4 v = ((const float4*)h)[s * 16 + q];
    float* a = agg + (size_t)d * HID + q * 4;
    atomicAdd(a + 0, v.x);
    atomicAdd(a + 1, v.y);
    atomicAdd(a + 2, v.z);
    atomicAdd(a + 3, v.w);
}

// ----------------------------------------------- WMMA SAGE linear (64 -> 64)
// out = relu( (agg/deg) @ Wl + h @ Wr + b ) + h     one wave = 16 nodes x 64 cols
__global__ __launch_bounds__(128) void k_sage_gemm(
        const float* __restrict__ agg, const float* __restrict__ deg,
        const float* __restrict__ hin,
        const unsigned short* __restrict__ wl_bf, const unsigned short* __restrict__ wr_bf,
        const float* __restrict__ bias, float* __restrict__ hout, int N)
{
    __shared__ __align__(16) unsigned short sWl[64 * 64];    // pre-swizzled B frags
    __shared__ __align__(16) unsigned short sWr[64 * 64];
    __shared__ __align__(16) unsigned short sM[4][16 * 64];  // per-wave mean tile [row][K]
    __shared__ __align__(16) unsigned short sH[4][16 * 64];  // per-wave h tile

    const int tid  = threadIdx.x;
    const int wid  = tid >> 5;
    const int lane = tid & 31;

    // weight staging: straight vector copy (layout already fragment-ordered)
    {
        const uint4* gl = (const uint4*)wl_bf;   // 512 uint4
        const uint4* gr = (const uint4*)wr_bf;
        uint4* dl = (uint4*)sWl;
        uint4* dr = (uint4*)sWr;
        #pragma unroll
        for (int i = 0; i < 4; ++i) {
            dl[tid + i * 128] = gl[tid + i * 128];
            dr[tid + i * 128] = gr[tid + i * 128];
        }
    }

    const int T     = (N + 15) >> 4;
    const int tile  = blockIdx.x * 4 + wid;
    const bool act  = (tile < T);
    const int node0 = tile * 16;
    const bool full = act && (node0 + 16 <= N);   // wave-uniform

    if (act) {
        // stage 16x64 mean (agg/deg) and h tiles into LDS as bf16 (packed b64 stores)
        if (full) {
            #pragma unroll
            for (int i = 0; i < 8; ++i) {
                int idx  = i * 32 + lane;        // 0..255
                int r    = idx >> 4;
                int c4   = idx & 15;
                int node = node0 + r;
                float inv = 1.0f / fmaxf(deg[node], 1.0f);
                float4 mv = ((const float4*)agg)[node * 16 + c4];
                float4 hv = ((const float4*)hin)[node * 16 + c4];
                int o = r * 64 + c4 * 4;
                *(uint2*)(&sM[wid][o]) = make_uint2(pack2(mv.x * inv, mv.y * inv),
                                                    pack2(mv.z * inv, mv.w * inv));
                *(uint2*)(&sH[wid][o]) = make_uint2(pack2(hv.x, hv.y),
                                                    pack2(hv.z, hv.w));
            }
        } else {
            for (int i = 0; i < 8; ++i) {
                int idx  = i * 32 + lane;
                int r    = idx >> 4;
                int c4   = idx & 15;
                int node = node0 + r;
                float4 mv = make_float4(0.f, 0.f, 0.f, 0.f);
                float4 hv = mv;
                float inv = 0.0f;
                if (node < N) {
                    inv = 1.0f / fmaxf(deg[node], 1.0f);
                    mv  = ((const float4*)agg)[node * 16 + c4];
                    hv  = ((const float4*)hin)[node * 16 + c4];
                }
                int o = r * 64 + c4 * 4;
                *(uint2*)(&sM[wid][o]) = make_uint2(pack2(mv.x * inv, mv.y * inv),
                                                    pack2(mv.z * inv, mv.w * inv));
                *(uint2*)(&sH[wid][o]) = make_uint2(pack2(hv.x, hv.y),
                                                    pack2(hv.z, hv.w));
            }
        }
    }
    __syncthreads();
    if (!act) return;

    const int half = lane >> 4;     // 0: lanes 0-15, 1: lanes 16-31
    const int l16  = lane & 15;

    // A fragments (depend only on k0): two contiguous b128 chunks per fragment
    v16bf aM[2], aH[2];
    {
        const int abase = l16 * 64;
        #pragma unroll
        for (int kk = 0; kk < 2; ++kk) {
            int b1 = abase + kk * 32 + half * 16;       // run K = k0+half*16 .. +7
            int b2 = b1 + 8;                            // run K = k0+half*16+8 .. +7
            Frag32B tm, th;
            tm.lo = *(const uint4*)(&sM[wid][b1]);
            tm.hi = *(const uint4*)(&sM[wid][b2]);
            th.lo = *(const uint4*)(&sH[wid][b1]);
            th.hi = *(const uint4*)(&sH[wid][b2]);
            aM[kk] = __builtin_bit_cast(v16bf, tm);
            aH[kk] = __builtin_bit_cast(v16bf, th);
        }
    }

    for (int nt = 0; nt < 4; ++nt) {
        const int n = nt * 16 + l16;                    // output column
        float bn = bias[n];
        v8f acc;
        #pragma unroll
        for (int v = 0; v < 8; ++v) acc[v] = bn;

        #pragma unroll
        for (int kk = 0; kk < 2; ++kk) {
            // B fragments: 32 contiguous bytes in swizzled LDS layout
            int wb = (((kk * 64 + n) * 2) + half) * 16;
            Frag32B tl, tr;
            tl.lo = *(const uint4*)(&sWl[wb]);
            tl.hi = *(const uint4*)(&sWl[wb + 8]);
            tr.lo = *(const uint4*)(&sWr[wb]);
            tr.hi = *(const uint4*)(&sWr[wb + 8]);
            v16bf bL = __builtin_bit_cast(v16bf, tl);
            v16bf bR = __builtin_bit_cast(v16bf, tr);
            acc = __builtin_amdgcn_wmma_f32_16x16x32_bf16(false, aM[kk], false, bL,
                                                          (short)0, acc, false, false);
            acc = __builtin_amdgcn_wmma_f32_16x16x32_bf16(false, aH[kk], false, bR,
                                                          (short)0, acc, false, false);
        }

        // epilogue: relu + residual ; D layout: M = half*8 + v, N = l16
        if (full) {
            #pragma unroll
            for (int v = 0; v < 8; ++v) {
                int idx = (node0 + half * 8 + v) * HID + n;
                float res = hin[idx];
                hout[idx] = fmaxf(acc[v], 0.0f) + res;
            }
        } else {
            #pragma unroll
            for (int v = 0; v < 8; ++v) {
                int node = node0 + half * 8 + v;
                if (node < N) {
                    int idx = node * HID + n;
                    float res = hin[idx];
                    hout[idx] = fmaxf(acc[v], 0.0f) + res;
                }
            }
        }
    }
}

// ------------------------------------------------------------------ pooling
__global__ void k_pool(const float* __restrict__ h, const int* __restrict__ batch,
                       float* __restrict__ pooled, float* __restrict__ cnt, int N) {
    int n = blockIdx.x;
    int c = threadIdx.x;
    if (n >= N) return;
    int g = batch[n];
    atomicAdd(&pooled[g * HID + c], h[n * HID + c]);
    if (c == 0) atomicAdd(&cnt[g], 1.0f);
}

// --------------------------------------------------------------------- head
__global__ void k_head(const float* __restrict__ pooled, const float* __restrict__ cnt,
                       const float* __restrict__ Wfc1, const float* __restrict__ bfc1,
                       const float* __restrict__ Wfc2, const float* __restrict__ bfc2,
                       float* __restrict__ out) {
    __shared__ float red[32];
    int g = blockIdx.x;
    int j = threadIdx.x;                 // 0..31
    float inv = 1.0f / fmaxf(cnt[g], 1.0f);
    float t = bfc1[j];
    for (int k = 0; k < HID; ++k)
        t += pooled[g * HID + k] * inv * Wfc1[k * 32 + j];
    t = fmaxf(t, 0.0f);
    red[j] = t * Wfc2[j];
    __syncthreads();
    if (j == 0) {
        float s = bfc2[0];
        for (int i = 0; i < 32; ++i) s += red[i];
        out[g] = s;
    }
}

// ===========================================================================
extern "C" void kernel_launch(void* const* d_in, const int* in_sizes, int n_in,
                              void* d_out, int out_size, void* d_ws, size_t ws_size,
                              hipStream_t stream) {
    const float* x    = (const float*)d_in[0];
    const int*   edge = (const int*)  d_in[1];
    const int*   batch= (const int*)  d_in[2];
    const float* W1l  = (const float*)d_in[3];
    const float* b1   = (const float*)d_in[4];
    const float* W1r  = (const float*)d_in[5];
    const float* W2l  = (const float*)d_in[6];
    const float* b2   = (const float*)d_in[7];
    const float* W2r  = (const float*)d_in[8];
    const float* W3l  = (const float*)d_in[9];
    const float* b3   = (const float*)d_in[10];
    const float* W3r  = (const float*)d_in[11];
    const float* Wfc1 = (const float*)d_in[12];
    const float* bfc1 = (const float*)d_in[13];
    const float* Wfc2 = (const float*)d_in[14];
    const float* bfc2 = (const float*)d_in[15];
    float* out = (float*)d_out;

    const int N = in_sizes[0];            // 50000 (x is [N,1])
    const int E = in_sizes[1] / 2;        // 800000
    const int G = out_size;               // 64 graphs
    const int* esrc = edge;
    const int* edst = edge + E;

    // ---- workspace carve-up (256B aligned)
    char* ws = (char*)d_ws;
    size_t off = 0;
    auto carve = [&](size_t bytes) -> void* {
        void* p = ws + off;
        off = (off + bytes + 255) & ~(size_t)255;
        return p;
    };
    float* hA     = (float*)carve((size_t)N * HID * sizeof(float));
    float* hB     = (float*)carve((size_t)N * HID * sizeof(float));
    float* agg    = (float*)carve((size_t)N * HID * sizeof(float));
    float* deg    = (float*)carve((size_t)N * sizeof(float));
    float* sx     = (float*)carve((size_t)N * sizeof(float));
    float* pooled = (float*)carve((size_t)G * HID * sizeof(float));
    float* cnt    = (float*)carve((size_t)G * sizeof(float));
    unsigned short* wb2l = (unsigned short*)carve(64 * 64 * sizeof(unsigned short));
    unsigned short* wb2r = (unsigned short*)carve(64 * 64 * sizeof(unsigned short));
    unsigned short* wb3l = (unsigned short*)carve(64 * 64 * sizeof(unsigned short));
    unsigned short* wb3r = (unsigned short*)carve(64 * 64 * sizeof(unsigned short));

    const int ZB = 256;
    // ---- init
    k_zero<<<(N + ZB - 1) / ZB, ZB, 0, stream>>>(deg, N);
    k_zero<<<(N + ZB - 1) / ZB, ZB, 0, stream>>>(sx, N);
    k_w_to_bf16_swz<<<16, 256, 0, stream>>>(W2l, wb2l, 64 * 64);
    k_w_to_bf16_swz<<<16, 256, 0, stream>>>(W2r, wb2r, 64 * 64);
    k_w_to_bf16_swz<<<16, 256, 0, stream>>>(W3l, wb3l, 64 * 64);
    k_w_to_bf16_swz<<<16, 256, 0, stream>>>(W3r, wb3r, 64 * 64);

    // ---- layer 1 (scalar features)
    k_edge_deg_sx<<<(E + ZB - 1) / ZB, ZB, 0, stream>>>(esrc, edst, x, sx, deg, E);
    k_layer1<<<N, HID, 0, stream>>>(x, sx, deg, W1l, b1, W1r, hA, N);

    const int T = (N + 15) / 16;
    const int gemm_blocks = (T + 3) / 4;
    const int agg_threads = E * 16;

    // ---- layer 2
    k_zero<<<((N * HID) + ZB - 1) / ZB, ZB, 0, stream>>>(agg, N * HID);
    k_aggregate<<<(agg_threads + ZB - 1) / ZB, ZB, 0, stream>>>(esrc, edst, hA, agg, E);
    k_sage_gemm<<<gemm_blocks, 128, 0, stream>>>(agg, deg, hA, wb2l, wb2r, b2, hB, N);

    // ---- layer 3
    k_zero<<<((N * HID) + ZB - 1) / ZB, ZB, 0, stream>>>(agg, N * HID);
    k_aggregate<<<(agg_threads + ZB - 1) / ZB, ZB, 0, stream>>>(esrc, edst, hB, agg, E);
    k_sage_gemm<<<gemm_blocks, 128, 0, stream>>>(agg, deg, hB, wb3l, wb3r, b3, hA, N);

    // ---- pool + head
    k_zero<<<((G * HID) + ZB - 1) / ZB, ZB, 0, stream>>>(pooled, G * HID);
    k_zero<<<1, ZB, 0, stream>>>(cnt, G);
    k_pool<<<N, HID, 0, stream>>>(hA, batch, pooled, cnt, N);
    k_head<<<G, 32, 0, stream>>>(pooled, cnt, Wfc1, bfc1, Wfc2, bfc2, out);
}
